// DiffusionPolicy_1563368095911
// MI455X (gfx1250) — matrix operand order repaired and checked by reading, compiled, not run on hardware
//
#include <hip/hip_runtime.h>
#include <math.h>

// ---------------------------------------------------------------------------
// MI455X (gfx1250) diffusion-policy UNet, 10 DDIM steps, B=4096.
// All GEMM-like work lowered to v_wmma_f32_16x16x32_f16 (wave32 WMMA).
// Activations: f32 master + f16 shadow, [T][B][C] layout (channel contiguous).
// GEMM B-fragments read the f16 shadow -> branch-free, convert-free hot loop.
// Weights: packed to f16 A-fragment layout once per launch (deterministic).
// ---------------------------------------------------------------------------

#define BATCH    4096
#define COND_DIM 192
#define TPRED    16
#define XPAD     32          // padded channel stride of the trajectory buffer

typedef _Float16 h8   __attribute__((ext_vector_type(8)));
typedef _Float16 v16h __attribute__((ext_vector_type(16)));
typedef float    v8f  __attribute__((ext_vector_type(8)));

__device__ __forceinline__ float mishf(float x) {
  float sp = (x > 20.f) ? x : log1pf(__expf(x));
  return x * tanhf(sp);
}

// ---------------- weight packing: [tap][co(pad16)][ci(pad32)] f16 ----------
__global__ void dp_pack_w(const float* __restrict__ w, _Float16* __restrict__ dst,
                          int Cout, int Cin, int taps, int copad, int cipad,
                          int transposed, long long total) {
  long long i = (long long)blockIdx.x * blockDim.x + threadIdx.x;
  if (i >= total) return;
  int ci = (int)(i % cipad);
  long long r = i / cipad;
  int co = (int)(r % copad);
  int tap = (int)(r / copad);
  float v = 0.f;
  if (co < Cout && ci < Cin)
    v = transposed ? w[((long long)ci * Cout + co) * taps + tap]
                   : w[((long long)co * Cin + ci) * taps + tap];
  dst[i] = (_Float16)v;
}

// ---------------- generic WMMA conv/linear over TBC activations ------------
// D(16 co x 16 batch) per wave.  A = weights (M=co, K=ci), B = acts (K=ci, N=b).
// tap loop: t_in = tout*a_str + b_off + ksign*k  (divided by `up` if up>1).
// Xh: f16 shadow of input (stride CinX == cipad, zero-padded).
// Writes f32 master Y and f16 shadow Yh.
__global__ __launch_bounds__(32) void dp_gemm_tbc(
    const _Float16* __restrict__ Xh, const _Float16* __restrict__ Wp,
    const float* __restrict__ bias, float* __restrict__ Y,
    _Float16* __restrict__ Yh,
    int Tin, int CinX, int CoutReal,
    int copad, int cipad, int taps,
    int a_str, int b_off, int ksign, int up) {
  const int lane = threadIdx.x;
  const int m15 = lane & 15;
  const int hi = lane >> 4;
  const int btile = blockIdx.x;
  const int cotile = blockIdx.y;
  const int tout = blockIdx.z;
  const int bidx = btile * 16 + m15;

  v8f acc = {0.f, 0.f, 0.f, 0.f, 0.f, 0.f, 0.f, 0.f};

  for (int k = 0; k < taps; ++k) {
    int tnum = tout * a_str + b_off + ksign * k;
    if (up > 1) {
      if (tnum < 0 || (tnum & (up - 1))) continue;
      tnum >>= 1;
    }
    if (tnum < 0 || tnum >= Tin) continue;
    const _Float16* xrow = Xh + ((size_t)tnum * BATCH + bidx) * CinX;
    const _Float16* wrow = Wp + ((size_t)k * copad + cotile * 16 + m15) * (size_t)cipad;
    for (int kk = 0; kk < cipad; kk += 32) {
      // A fragment (weights): lane<16 -> K=kk+0..7,16..23 ; lane>=16 -> +8
      union { v16h v; h8 h[2]; } A;
      A.h[0] = *(const h8*)(wrow + kk + hi * 8);
      A.h[1] = *(const h8*)(wrow + kk + 16 + hi * 8);
      // B fragment (activations): lane<16 -> K=kk+0..7,16..23 ; lane>=16 -> +8
      union { v16h v; h8 h[2]; } B;
      B.h[0] = *(const h8*)(xrow + kk + hi * 8);
      B.h[1] = *(const h8*)(xrow + kk + 16 + hi * 8);
      acc = __builtin_amdgcn_wmma_f32_16x16x32_f16(
          false, A.v, false, B.v, (short)0, acc, false, false);
    }
  }

  const int cobase = cotile * 16 + hi * 8;
  float* yrow = Y + ((size_t)tout * BATCH + bidx) * CoutReal;
  _Float16* yhrow = Yh + ((size_t)tout * BATCH + bidx) * CoutReal;
  if (cobase + 8 <= CoutReal) {
    h8 hv;
#pragma unroll
    for (int r = 0; r < 8; ++r) {
      float v = acc[r] + bias[cobase + r];
      yrow[cobase + r] = v;
      hv[r] = (_Float16)v;
    }
    *(h8*)(yhrow + cobase) = hv;
  } else {
#pragma unroll
    for (int r = 0; r < 8; ++r) {
      int co = cobase + r;
      if (co < CoutReal) {
        float v = acc[r] + bias[co];
        yrow[co] = v;
        yhrow[co] = (_Float16)v;
      }
    }
  }
}

// ---------------- GroupNorm(8 groups) + Mish, always 128 elems/group -------
__global__ __launch_bounds__(128) void dp_gn_mish(
    float* __restrict__ Y, _Float16* __restrict__ Yh,
    const float* __restrict__ gs, const float* __restrict__ gb, int C) {
  int bg = blockIdx.x;
  int b = bg >> 3, g = bg & 7;
  int cg = C >> 3;                       // channels per group; T*cg == 128
  int e = threadIdx.x;
  int t = e / cg, ci = e - t * cg;
  int c = g * cg + ci;
  size_t idx = ((size_t)t * BATCH + b) * C + c;
  float v = Y[idx];
  float s = v, q = v * v;
  for (int o = 16; o > 0; o >>= 1) {
    s += __shfl_xor(s, o, 32);
    q += __shfl_xor(q, o, 32);
  }
  __shared__ float sh1[4], sh2[4];
  int wid = threadIdx.x >> 5;
  if ((threadIdx.x & 31) == 0) { sh1[wid] = s; sh2[wid] = q; }
  __syncthreads();
  float sum = sh1[0] + sh1[1] + sh1[2] + sh1[3];
  float sq = sh2[0] + sh2[1] + sh2[2] + sh2[3];
  float mean = sum * (1.f / 128.f);
  float var = sq * (1.f / 128.f) - mean * mean;
  float r = rsqrtf(var + 1e-5f);
  float xn = (v - mean) * r * gs[c] + gb[c];
  float o = mishf(xn);
  Y[idx] = o;
  Yh[idx] = (_Float16)o;
}

// ---------------- elementwise kernels --------------------------------------
__global__ void dp_film(float* __restrict__ Y, _Float16* __restrict__ Yh,
                        const float* __restrict__ emb, int C, long long total) {
  long long i = (long long)blockIdx.x * blockDim.x + threadIdx.x;
  if (i >= total) return;
  int c = (int)(i % C);
  long long tb = i / C;
  int b = (int)(tb % BATCH);
  size_t e = (size_t)b * 2 * C;
  float v = emb[e + c] * Y[i] + emb[e + C + c];
  Y[i] = v;
  Yh[i] = (_Float16)v;
}

__global__ void dp_add(float* __restrict__ Y, _Float16* __restrict__ Yh,
                       const float* __restrict__ R, long long total) {
  long long i = (long long)blockIdx.x * blockDim.x + threadIdx.x;
  if (i >= total) return;
  float v = Y[i] + R[i];
  Y[i] = v;
  Yh[i] = (_Float16)v;
}

__global__ void dp_concat(const float* __restrict__ X, const float* __restrict__ S,
                          float* __restrict__ O, _Float16* __restrict__ Oh,
                          int C1, int C2, long long total) {
  long long i = (long long)blockIdx.x * blockDim.x + threadIdx.x;
  if (i >= total) return;
  int C = C1 + C2;
  int c = (int)(i % C);
  long long tb = i / C;
  float v = (c < C1) ? X[tb * C1 + c] : S[tb * C2 + (c - C1)];
  O[i] = v;
  Oh[i] = (_Float16)v;
}

// noise (B,16,8) -> padded TBC [t][b][XPAD]; channels 8..31 zeroed
__global__ void dp_noise2tbc(const float* __restrict__ nz, float* __restrict__ x,
                             _Float16* __restrict__ xh, long long total) {
  long long i = (long long)blockIdx.x * blockDim.x + threadIdx.x;
  if (i >= total) return;                 // total = B*16*XPAD
  int c = (int)(i & (XPAD - 1));
  long long tb = i >> 5;
  int b = (int)(tb % BATCH);
  int t = (int)(tb / BATCH);
  float v = (c < 8) ? nz[((size_t)b * TPRED + t) * 8 + c] : 0.f;
  x[i] = v;
  xh[i] = (_Float16)v;
}

__global__ void dp_mcond_obs(const float* __restrict__ obs, float* __restrict__ mcond,
                             _Float16* __restrict__ mcondh, long long total) {
  long long i = (long long)blockIdx.x * blockDim.x + threadIdx.x;
  if (i >= total) return;                 // total = B*128
  int b = (int)(i >> 7);
  int j = (int)(i & 127);
  float v = mishf(obs[i]);
  size_t o = (size_t)b * COND_DIM + 64 + j;
  mcond[o] = v;
  mcondh[o] = (_Float16)v;
}

__global__ void dp_mcond_temb(const float* __restrict__ temb, float* __restrict__ mcond,
                              _Float16* __restrict__ mcondh, long long total) {
  long long i = (long long)blockIdx.x * blockDim.x + threadIdx.x;
  if (i >= total) return;                 // total = B*64
  int b = (int)(i >> 6);
  int c = (int)(i & 63);
  float v = mishf(temb[c]);
  size_t o = (size_t)b * COND_DIM + c;
  mcond[o] = v;
  mcondh[o] = (_Float16)v;
}

__global__ __launch_bounds__(256) void dp_temb(
    const float* __restrict__ l1w, const float* __restrict__ l1b,
    const float* __restrict__ l2w, const float* __restrict__ l2b,
    float tval, float* __restrict__ temb) {
  __shared__ float emb[64];
  __shared__ float h1[256];
  int tid = threadIdx.x;
  if (tid < 32) {
    float fr = __expf(-logf(10000.f) / 31.f * (float)tid);
    float arg = tval * fr;
    emb[tid] = sinf(arg);
    emb[tid + 32] = cosf(arg);
  }
  __syncthreads();
  float h = l1b[tid];
  for (int j = 0; j < 64; ++j) h += emb[j] * l1w[tid * 64 + j];
  h1[tid] = mishf(h);
  __syncthreads();
  if (tid < 64) {
    float o = l2b[tid];
    for (int i = 0; i < 256; ++i) o += h1[i] * l2w[tid * 256 + i];
    temb[tid] = o;
  }
}

// DDIM update on the 8 real channels of the padded trajectory buffer.
__global__ void dp_ddim(float* __restrict__ x, _Float16* __restrict__ xh,
                        const float* __restrict__ eps,
                        float inv_s0, float c1, float sp, float se,
                        long long total) {
  long long i = (long long)blockIdx.x * blockDim.x + threadIdx.x;
  if (i >= total) return;                 // total = B*16*8  (i = (t*B+b)*8+c)
  int c = (int)(i & 7);
  long long tb = i >> 3;
  size_t xi = (size_t)tb * XPAD + c;
  float e = eps[i];
  float x0 = (x[xi] - c1 * e) * inv_s0;
  x0 = fminf(1.f, fmaxf(-1.f, x0));
  float v = sp * x0 + se * e;
  x[xi] = v;
  xh[xi] = (_Float16)v;
}

__global__ void dp_final(const float* __restrict__ x, float* __restrict__ out,
                         long long total) {
  long long i = (long long)blockIdx.x * blockDim.x + threadIdx.x;
  if (i >= total) return;                 // total = B*8*8, out[b][th][c]
  int b = (int)(i >> 6);
  int r = (int)(i & 63);
  int th = r >> 3, c = r & 7;
  out[i] = x[((size_t)(th + 1) * BATCH + b) * XPAD + c];  // start = OBS_H-1 = 1
}

// ===========================================================================
extern "C" void kernel_launch(void* const* d_in, const int* in_sizes, int n_in,
                              void* d_out, int out_size, void* d_ws, size_t ws_size,
                              hipStream_t stream) {
  (void)in_sizes; (void)n_in; (void)out_size; (void)ws_size;
  const float* obs_seq = (const float*)d_in[0];       // (B, 2, 64)
  const float* init_noise = (const float*)d_in[1];    // (B, 16, 8)
  float* out = (float*)d_out;

  // ---- workspace carve-up -------------------------------------------------
  _Float16* WP = (_Float16*)d_ws;         // packed f16 weights (< 16 MB)
  size_t off = (size_t)16 << 20;
  auto carve = [&](size_t bytes) -> void* {
    void* p = (char*)d_ws + off;
    off = (off + bytes + 255) & ~(size_t)255;
    return p;
  };
  struct Buf { float* f; _Float16* h; };
  auto allocBuf = [&](size_t elems) -> Buf {
    Buf b;
    b.f = (float*)carve(elems * 4);
    b.h = (_Float16*)carve(elems * 2);
    return b;
  };
  Buf mcond = allocBuf((size_t)BATCH * COND_DIM);     // mish(cond), [B][192]
  float* temb = (float*)carve(4096);
  Buf embB  = allocBuf((size_t)BATCH * 512);          // FiLM emb, [B][2C]
  Buf xtraj = allocBuf((size_t)BATCH * TPRED * XPAD); // sample, padded TBC
  Buf yout  = allocBuf((size_t)BATCH * TPRED * 8);    // eps, TBC stride 8
  Buf a0 = allocBuf((size_t)BATCH * 1024);
  Buf a1 = allocBuf((size_t)BATCH * 1024);
  Buf a2 = allocBuf((size_t)BATCH * 1024);
  Buf sk1 = allocBuf((size_t)BATCH * 1024);           // d1 skip (128ch, T=8)
  Buf sk2 = allocBuf((size_t)BATCH * 1024);           // d2 skip (256ch, T=4)
  Buf catb = allocBuf((size_t)BATCH * 2048);          // concat buffer

  // ---- param walk in dict-insertion order + f16 packing -------------------
  int pi = 2;
  auto nxt = [&]() -> const float* { return (const float*)d_in[pi++]; };

  struct Packed { size_t offh; int copad, cipad, taps; };
  size_t wpH = 0;
  auto packW = [&](const float* w, int Cout, int Cin, int taps, bool tr) -> Packed {
    Packed p;
    p.copad = (Cout + 15) & ~15;
    p.cipad = (Cin + 31) & ~31;
    p.taps = taps;
    p.offh = wpH;
    long long tot = (long long)taps * p.copad * p.cipad;
    wpH += ((size_t)tot + 127) & ~(size_t)127;
    dp_pack_w<<<dim3((unsigned)((tot + 255) / 256)), 256, 0, stream>>>(
        w, WP + p.offh, Cout, Cin, taps, p.copad, p.cipad, tr ? 1 : 0, tot);
    return p;
  };

  const float* tl1w = nxt(); const float* tl1b = nxt();
  const float* tl2w = nxt(); const float* tl2b = nxt();

  struct CB { Packed w; const float *b, *gs, *gb; int Cin, Cout; };
  auto addCB = [&](int cin, int cout, int k) -> CB {
    CB c; const float* w = nxt();
    c.b = nxt(); c.gs = nxt(); c.gb = nxt();
    c.w = packW(w, cout, cin, k, false);
    c.Cin = cin; c.Cout = cout;
    return c;
  };
  struct RB { CB b0, b1; Packed cw; const float* cb; bool hasR; Packed rw;
              const float* rb; int Cin, Cout; };
  auto addRB = [&](int cin, int cout) -> RB {
    RB r; r.b0 = addCB(cin, cout, 5); r.b1 = addCB(cout, cout, 5);
    const float* cw = nxt();
    r.cw = packW(cw, 2 * cout, COND_DIM, 1, false);
    r.cb = nxt();
    r.hasR = (cin != cout);
    if (r.hasR) { const float* rw = nxt(); r.rw = packW(rw, cout, cin, 1, false); r.rb = nxt(); }
    r.Cin = cin; r.Cout = cout;
    return r;
  };

  RB d0r0 = addRB(8, 64),    d0r1 = addRB(64, 64);
  Packed ds0 = packW(nxt(), 64, 64, 3, false);  const float* ds0b = nxt();
  RB d1r0 = addRB(64, 128),  d1r1 = addRB(128, 128);
  Packed ds1 = packW(nxt(), 128, 128, 3, false); const float* ds1b = nxt();
  RB d2r0 = addRB(128, 256), d2r1 = addRB(256, 256);
  RB m0 = addRB(256, 256),   m1 = addRB(256, 256);
  RB u0r0 = addRB(512, 128), u0r1 = addRB(128, 128);
  Packed us0 = packW(nxt(), 128, 128, 4, true);  const float* us0b = nxt();
  RB u1r0 = addRB(256, 64),  u1r1 = addRB(64, 64);
  Packed us1 = packW(nxt(), 64, 64, 4, true);    const float* us1b = nxt();
  CB f0 = addCB(64, 64, 5);
  Packed f1p = packW(nxt(), 8, 64, 1, false);    const float* f1b = nxt();

  // ---- launch helpers -----------------------------------------------------
  auto gemm = [&](const _Float16* Xh, const Packed& W, const float* bias, Buf Y,
                  int Tin, int Tout, int CinX, int CoutReal,
                  int a_str, int b_off, int ksign, int up) {
    dim3 g(BATCH / 16, W.copad / 16, Tout);
    dp_gemm_tbc<<<g, 32, 0, stream>>>(Xh, WP + W.offh, bias, Y.f, Y.h,
                                      Tin, CinX, CoutReal, W.copad, W.cipad,
                                      W.taps, a_str, b_off, ksign, up);
  };
  auto ew = [&](long long n) { return dim3((unsigned)((n + 255) / 256)); };

  // CinX = channel stride of xin buffer (== cipad of the consuming layer)
  auto run_res = [&](const RB& r, Buf xin, int CinX, int T, Buf h, Buf outb) {
    long long tot = (long long)T * BATCH * r.Cout;
    gemm(xin.h, r.b0.w, r.b0.b, h, T, T, CinX, r.Cout, 1, -2, 1, 1);
    dp_gn_mish<<<BATCH * 8, 128, 0, stream>>>(h.f, h.h, r.b0.gs, r.b0.gb, r.Cout);
    gemm(mcond.h, r.cw, r.cb, embB, 1, 1, COND_DIM, 2 * r.Cout, 1, 0, 1, 1);
    dp_film<<<ew(tot), 256, 0, stream>>>(h.f, h.h, embB.f, r.Cout, tot);
    gemm(h.h, r.b1.w, r.b1.b, outb, T, T, r.Cout, r.Cout, 1, -2, 1, 1);
    dp_gn_mish<<<BATCH * 8, 128, 0, stream>>>(outb.f, outb.h, r.b1.gs, r.b1.gb, r.Cout);
    if (r.hasR) {
      gemm(xin.h, r.rw, r.rb, h, T, T, CinX, r.Cout, 1, 0, 1, 1);
      dp_add<<<ew(tot), 256, 0, stream>>>(outb.f, outb.h, h.f, tot);
    } else {
      dp_add<<<ew(tot), 256, 0, stream>>>(outb.f, outb.h, xin.f, tot);
    }
  };

  // ---- DDIM schedule (squaredcos_cap_v2), host double math ----------------
  double AC[100];
  {
    auto ab = [](double t) {
      double v = cos((t + 0.008) / 1.008 * 3.14159265358979323846 / 2.0);
      return v * v;
    };
    double prod = 1.0;
    for (int i = 0; i < 100; ++i) {
      double beta = 1.0 - ab((i + 1) / 100.0) / ab(i / 100.0);
      if (beta > 0.999) beta = 0.999;
      prod *= (1.0 - beta);
      AC[i] = prod;
    }
  }

  // ---- static per-call setup ---------------------------------------------
  dp_noise2tbc<<<ew((long long)BATCH * TPRED * XPAD), 256, 0, stream>>>(
      init_noise, xtraj.f, xtraj.h, (long long)BATCH * TPRED * XPAD);
  dp_mcond_obs<<<ew((long long)BATCH * 128), 256, 0, stream>>>(
      obs_seq, mcond.f, mcond.h, (long long)BATCH * 128);

  // ---- 10 DDIM steps ------------------------------------------------------
  for (int s = 0; s < 10; ++s) {
    int t = 90 - s * 10;
    dp_temb<<<1, 256, 0, stream>>>(tl1w, tl1b, tl2w, tl2b, (float)t, temb);
    dp_mcond_temb<<<ew((long long)BATCH * 64), 256, 0, stream>>>(
        temb, mcond.f, mcond.h, (long long)BATCH * 64);

    // -------- UNet forward --------
    run_res(d0r0, xtraj, XPAD, 16, a0, a1);
    run_res(d0r1, a1, 64, 16, a0, a2);
    gemm(a2.h, ds0, ds0b, a0, 16, 8, 64, 64, 2, -1, 1, 1);          // down 0
    run_res(d1r0, a0, 64, 8, a1, a2);
    run_res(d1r1, a2, 128, 8, a1, sk1);                              // skip d1
    gemm(sk1.h, ds1, ds1b, a0, 8, 4, 128, 128, 2, -1, 1, 1);         // down 1
    run_res(d2r0, a0, 128, 4, a1, a2);
    run_res(d2r1, a2, 256, 4, a1, sk2);                              // skip d2
    run_res(m0, sk2, 256, 4, a1, a0);
    run_res(m1, a0, 256, 4, a1, a2);
    // up 0: cat(x, d2 skip) -> 512ch, T=4
    {
      long long tot = (long long)4 * BATCH * 512;
      dp_concat<<<ew(tot), 256, 0, stream>>>(a2.f, sk2.f, catb.f, catb.h,
                                             256, 256, tot);
    }
    run_res(u0r0, catb, 512, 4, a0, a1);
    run_res(u0r1, a1, 128, 4, a0, a2);
    gemm(a2.h, us0, us0b, a0, 4, 8, 128, 128, 1, 1, -1, 2);          // up-conv 0
    // up 1: cat(x, d1 skip) -> 256ch, T=8
    {
      long long tot = (long long)8 * BATCH * 256;
      dp_concat<<<ew(tot), 256, 0, stream>>>(a0.f, sk1.f, catb.f, catb.h,
                                             128, 128, tot);
    }
    run_res(u1r0, catb, 256, 8, a1, a2);
    run_res(u1r1, a2, 64, 8, a1, a0);
    gemm(a0.h, us1, us1b, a1, 8, 16, 64, 64, 1, 1, -1, 2);           // up-conv 1
    gemm(a1.h, f0.w, f0.b, a2, 16, 16, 64, 64, 1, -2, 1, 1);         // final block
    dp_gn_mish<<<BATCH * 8, 128, 0, stream>>>(a2.f, a2.h, f0.gs, f0.gb, 64);
    gemm(a2.h, f1p, f1b, yout, 16, 16, 64, 8, 1, 0, 1, 1);           // 1x1 head

    // -------- DDIM update --------
    double a_t = AC[t];
    int prev = t - 10;
    double a_prev = (prev >= 0) ? AC[prev] : 1.0;
    float inv_s0 = (float)(1.0 / sqrt(a_t));
    float c1 = (float)sqrt(1.0 - a_t);
    float sp = (float)sqrt(a_prev);
    float se = (float)sqrt(1.0 - a_prev);
    long long tot = (long long)BATCH * TPRED * 8;
    dp_ddim<<<ew(tot), 256, 0, stream>>>(xtraj.f, xtraj.h, yout.f,
                                         inv_s0, c1, sp, se, tot);
  }

  // ---- slice traj[:, 1:9, :] ---------------------------------------------
  dp_final<<<ew((long long)BATCH * 64), 256, 0, stream>>>(
      xtraj.f, out, (long long)BATCH * 64);
}
